// GlobalMatchingTokenizer_20151986553457
// MI455X (gfx1250) — matrix operand order repaired and checked by reading, not compiled, not run on hardware
//
#include <hip/hip_runtime.h>
#include <math.h>

typedef __attribute__((ext_vector_type(16))) __bf16 v16bf;
typedef __attribute__((ext_vector_type(8)))  __bf16 v8bf;
typedef __attribute__((ext_vector_type(8)))  float  v8f;

#define TOKEN_DIM 192
#define NHEADS 4
#define HEAD_DIM 48
#define TOPK 128
#define AWAVES 8   // waves per attention block

// monotone float -> uint key (ascending order preserved)
__device__ __forceinline__ unsigned sortkey(float f) {
  unsigned u = __float_as_uint(f);
  return (u & 0x80000000u) ? ~u : (u | 0x80000000u);
}

// intra-wave LDS ordering: CDNA5 split dep counter wait
__device__ __forceinline__ void wave_lds_sync() {
  __builtin_amdgcn_wave_barrier();
  asm volatile("s_wait_dscnt 0" ::: "memory");
  __builtin_amdgcn_wave_barrier();
}

// 2D sinusoidal positional encoding, token dim 192, dh = 96
__device__ __forceinline__ float pe_val(int token, int col, int Wimg) {
  float x = (float)(token % Wimg);
  float y = (float)(token / Wimg);
  const float c = -0.09594104554f; // -ln(10000)/96
  if (col < 96) return sinf(x * __expf(c * (float)col));
  return cosf(y * __expf(c * (float)(col - 96)));
}

// ---------------------------------------------------------------------------
// QKV projection: out[token, col] = sum_c feat[b, c, token] * w[c, col] + bias
// mode 0/1 : + PE, store bf16 into [B][4][N][64] (rows zero-padded 48..63)
// mode 2   : store f32 into [B][4][N][48]
// grid (N/16, 12, B), block 32 (1 wave per 16x16 tile)
// ---------------------------------------------------------------------------
__global__ void proj_kernel(const float* __restrict__ feat, const float* __restrict__ wmat,
                            const float* __restrict__ bias, void* __restrict__ outp,
                            int N, int Cin, int Wimg, int mode) {
  int lane = threadIdx.x & 31;
  int hi   = lane >> 4;      // 0|1
  int mcol = lane & 15;
  int tTile = blockIdx.x, cTile = blockIdx.y, b = blockIdx.z;
  int col   = cTile * 16 + mcol;   // output column for B/D fragments
  int token = tTile * 16 + mcol;   // A-fragment row (M = lane&15)

  const size_t fb = (size_t)b * Cin * N;
  int nchunk = Cin >> 5;           // C = 64/128/192 -> 2/4/6
  v8f acc = {0.f, 0.f, 0.f, 0.f, 0.f, 0.f, 0.f, 0.f};
  for (int ch = 0; ch < nchunk; ++ch) {
    v16bf a, bm;
    int ab0 = ch * 32 + (hi ? 16 : 0);
    int ab1 = ch * 32 + (hi ? 24 : 8);
#pragma unroll
    for (int e = 0; e < 8; ++e) {
      a[e]     = (__bf16)feat[fb + (size_t)(ab0 + e) * N + token];
      a[8 + e] = (__bf16)feat[fb + (size_t)(ab1 + e) * N + token];
    }
    int kb = ch * 32 + (hi ? 16 : 0);
#pragma unroll
    for (int e = 0; e < 16; ++e)
      bm[e] = (__bf16)wmat[(size_t)(kb + e) * TOKEN_DIM + col];
    acc = __builtin_amdgcn_wmma_f32_16x16x32_bf16(false, a, false, bm, (short)0,
                                                  acc, false, false);
  }
#pragma unroll
  for (int v = 0; v < 8; ++v) {
    int m   = v + (hi ? 8 : 0);
    int tok = tTile * 16 + m;
    float val = acc[v] + bias[col];
    int h = col / HEAD_DIM, kd = col % HEAD_DIM;
    if (mode == 2) {
      float* V = (float*)outp;
      V[(((size_t)b * NHEADS + h) * N + tok) * HEAD_DIM + kd] = val;
    } else {
      val += pe_val(tok, col, Wimg);
      __bf16* Q = (__bf16*)outp;
      Q[(((size_t)b * NHEADS + h) * N + tok) * 64 + kd] = (__bf16)val;
    }
  }
}

// ---------------------------------------------------------------------------
// Attention: per (b, head, 16-query strip): WMMA score strip into LDS,
// exact top-128 threshold via radix-256 select over sortable keys,
// softmax over selected, weighted V gather. Out bf16 [B][N][192].
// grid (N/16, 4, B), block 256 (8 waves)
// ---------------------------------------------------------------------------
__global__ void attn_kernel(const __bf16* __restrict__ Qb, const __bf16* __restrict__ Kb,
                            const float* __restrict__ V, __bf16* __restrict__ attnOut,
                            int N, float scale) {
  extern __shared__ char smemRaw[];
  float* scores = (float*)smemRaw;                        // [16][N]
  int*   selIdx = (int*)(scores + (size_t)16 * N);        // [AWAVES][128]
  float* selW   = (float*)(selIdx + AWAVES * TOPK);       // [AWAVES][128]
  int*   selCnt = (int*)(selW + AWAVES * TOPK);           // [AWAVES]
  int*   histAll= (int*)(selCnt + AWAVES);                // [AWAVES][256]

  int lane = threadIdx.x & 31;
  int wv   = threadIdx.x >> 5;   // 0..7
  int hi   = lane >> 4;
  int mcol = lane & 15;
  int qTile = blockIdx.x, head = blockIdx.y, b = blockIdx.z;
  const size_t bh = (size_t)b * NHEADS + head;

  // ---- A fragments: Q strip rows qTile*16 .. +15 (K=48 padded to 64) ----
  const __bf16* Qrow = Qb + (bh * N + (size_t)qTile * 16) * 64;
  v16bf a0, a1;
  {
    int ab0 = hi ? 16 : 0, ab1 = hi ? 24 : 8;
    v8bf l0 = *(const v8bf*)(Qrow + (size_t)mcol * 64 + ab0);
    v8bf h0 = *(const v8bf*)(Qrow + (size_t)mcol * 64 + ab1);
    v8bf l1 = *(const v8bf*)(Qrow + (size_t)mcol * 64 + 32 + ab0);
    v8bf h1 = *(const v8bf*)(Qrow + (size_t)mcol * 64 + 32 + ab1);
    a0 = __builtin_shufflevector(l0, h0, 0,1,2,3,4,5,6,7,8,9,10,11,12,13,14,15);
    a1 = __builtin_shufflevector(l1, h1, 0,1,2,3,4,5,6,7,8,9,10,11,12,13,14,15);
  }

  // ---- score strip via WMMA ----
  int ntiles = N >> 4;
  int kb = hi ? 16 : 0;
  for (int t = wv; t < ntiles; t += AWAVES) {
    const __bf16* Krow = Kb + (bh * N + (size_t)t * 16) * 64;
    if (t + AWAVES < ntiles)
      __builtin_prefetch(Kb + (bh * N + (size_t)(t + AWAVES) * 16) * 64 +
                         (size_t)mcol * 64, 0, 1);
    v16bf b0 = *(const v16bf*)(Krow + (size_t)mcol * 64 + kb);
    v16bf b1 = *(const v16bf*)(Krow + (size_t)mcol * 64 + 32 + kb);
    v8f c = {0.f, 0.f, 0.f, 0.f, 0.f, 0.f, 0.f, 0.f};
    c = __builtin_amdgcn_wmma_f32_16x16x32_bf16(false, a0, false, b0, (short)0, c, false, false);
    c = __builtin_amdgcn_wmma_f32_16x16x32_bf16(false, a1, false, b1, (short)0, c, false, false);
#pragma unroll
    for (int v = 0; v < 8; ++v)
      scores[(size_t)(v + (hi ? 8 : 0)) * N + t * 16 + mcol] = c[v] * scale;
  }
  __syncthreads();

  int* hist = histAll + wv * 256;

  // ---- each wave owns 2 query rows ----
  for (int rr = 0; rr < 16 / AWAVES; ++rr) {
    int r = wv * (16 / AWAVES) + rr;
    const float* srow = scores + (size_t)r * N;

    // row max (float2 reads: all 64 LDS banks)
    float m = -3.4e38f;
    for (int i = lane * 2; i < N; i += 64) {
      float2 s2 = *(const float2*)(srow + i);
      m = fmaxf(m, fmaxf(s2.x, s2.y));
    }
#pragma unroll
    for (int off = 16; off > 0; off >>= 1) m = fmaxf(m, __shfl_xor(m, off, 32));

    // ---- radix-256 select: threshold = 128th largest sortable key ----
    unsigned prefix = 0;
    int remaining = TOPK;
#pragma unroll
    for (int byte = 3; byte >= 0; --byte) {
      for (int i = lane; i < 256; i += 32) hist[i] = 0;
      wave_lds_sync();
      for (int i = lane * 2; i < N; i += 64) {
        float2 s2 = *(const float2*)(srow + i);
        unsigned k0 = sortkey(s2.x), k1 = sortkey(s2.y);
        if (byte == 3 || (k0 >> ((byte + 1) * 8)) == prefix)
          atomicAdd(&hist[(k0 >> (byte * 8)) & 255], 1);
        if (byte == 3 || (k1 >> ((byte + 1) * 8)) == prefix)
          atomicAdd(&hist[(k1 >> (byte * 8)) & 255], 1);
      }
      wave_lds_sync();
      // wave-parallel descending scan over 256 bins (8 groups of 32)
      int t = -1, cumBefore = 0;
      for (int g = 0; g < 8; ++g) {
        int bin = 255 - 32 * g - lane;
        int c = hist[bin];
        int p = c;
#pragma unroll
        for (int off = 1; off < 32; off <<= 1) {
          int nb = __shfl_up(p, off, 32);
          if (lane >= off) p += nb;
        }
        int total = __shfl(p, 31, 32);
        if (cumBefore + total >= remaining) {
          unsigned long long mask = __ballot(cumBefore + p >= remaining);
          int first = __ffsll(mask) - 1;
          t = 255 - 32 * g - first;
          int above = cumBefore + __shfl(p, first, 32) - __shfl(c, first, 32);
          remaining -= above;
          break;
        }
        cumBefore += total;
      }
      prefix = (prefix << 8) | (unsigned)t;
      wave_lds_sync();
    }
    unsigned thresh = prefix;

    // ---- compact selected (idx, weight) into per-wave LDS list ----
    if (lane == 0) selCnt[wv] = 0;
    wave_lds_sync();
    for (int i = lane * 2; i < N; i += 64) {
      float2 s2 = *(const float2*)(srow + i);
      if (sortkey(s2.x) >= thresh) {
        int pos = atomicAdd(&selCnt[wv], 1);
        if (pos < TOPK) { selIdx[wv * TOPK + pos] = i;     selW[wv * TOPK + pos] = __expf(s2.x - m); }
      }
      if (sortkey(s2.y) >= thresh) {
        int pos = atomicAdd(&selCnt[wv], 1);
        if (pos < TOPK) { selIdx[wv * TOPK + pos] = i + 1; selW[wv * TOPK + pos] = __expf(s2.y - m); }
      }
    }
    wave_lds_sync();
    int nsel = selCnt[wv]; if (nsel > TOPK) nsel = TOPK;

    float ws = 0.f;
    for (int e = lane; e < nsel; e += 32) ws += selW[wv * TOPK + e];
#pragma unroll
    for (int off = 16; off > 0; off >>= 1) ws += __shfl_xor(ws, off, 32);
    float inv = 1.f / ws;

    const float* Vb = V + bh * N * HEAD_DIM;
    for (int d = lane; d < HEAD_DIM; d += 32) {
      float acc = 0.f;
      for (int e = 0; e < nsel; ++e)
        acc += selW[wv * TOPK + e] * Vb[(size_t)selIdx[wv * TOPK + e] * HEAD_DIM + d];
      attnOut[((size_t)b * N + qTile * 16 + r) * TOKEN_DIM + head * HEAD_DIM + d] =
          (__bf16)(acc * inv);
    }
  }
}

// ---------------------------------------------------------------------------
// Output projection: out[b, col, token] = sum_j attnOut[b, token, j]*ow[j, col] + ob
// grid (N/16, 12, B), block 32
// ---------------------------------------------------------------------------
__global__ void outproj_kernel(const __bf16* __restrict__ A, const float* __restrict__ ow,
                               const float* __restrict__ ob, float* __restrict__ out, int N) {
  int lane = threadIdx.x & 31;
  int hi = lane >> 4, mcol = lane & 15;
  int tTile = blockIdx.x, cTile = blockIdx.y, b = blockIdx.z;
  int col   = cTile * 16 + mcol;
  int token = tTile * 16 + mcol;
  const __bf16* Ab = A + (size_t)b * N * TOKEN_DIM;

  v8f acc = {0.f, 0.f, 0.f, 0.f, 0.f, 0.f, 0.f, 0.f};
  for (int ch = 0; ch < 6; ++ch) {
    int ab0 = ch * 32 + (hi ? 16 : 0);
    int ab1 = ch * 32 + (hi ? 24 : 8);
    v8bf lo = *(const v8bf*)(Ab + (size_t)token * TOKEN_DIM + ab0);
    v8bf hh = *(const v8bf*)(Ab + (size_t)token * TOKEN_DIM + ab1);
    v16bf a = __builtin_shufflevector(lo, hh, 0,1,2,3,4,5,6,7,8,9,10,11,12,13,14,15);
    v16bf bm;
    int kb = ch * 32 + (hi ? 16 : 0);
#pragma unroll
    for (int e = 0; e < 16; ++e)
      bm[e] = (__bf16)ow[(size_t)(kb + e) * TOKEN_DIM + col];
    acc = __builtin_amdgcn_wmma_f32_16x16x32_bf16(false, a, false, bm, (short)0,
                                                  acc, false, false);
  }
#pragma unroll
  for (int v = 0; v < 8; ++v) {
    int m = v + (hi ? 8 : 0);
    int tok = tTile * 16 + m;
    out[((size_t)b * TOKEN_DIM + col) * N + tok] = acc[v] + ob[col];
  }
}

// ---------------------------------------------------------------------------
extern "C" void kernel_launch(void* const* d_in, const int* in_sizes, int n_in,
                              void* d_out, int out_size, void* d_ws, size_t ws_size,
                              hipStream_t stream) {
  (void)in_sizes; (void)n_in; (void)out_size; (void)ws_size;
  const int Cs[3] = {64, 128, 192};
  const int Hs[3] = {64, 32, 16};
  const int B = 2;
  float* outPtr = (float*)d_out;

  for (int l = 0; l < 3; ++l) {
    int C = Cs[l], H = Hs[l], W = Hs[l], N = H * W;
    const float* feat1 = (const float*)d_in[2 * l];
    const float* feat2 = (const float*)d_in[2 * l + 1];
    const float* qw = (const float*)d_in[6 + 8 * l + 0];
    const float* qb = (const float*)d_in[6 + 8 * l + 1];
    const float* kw = (const float*)d_in[6 + 8 * l + 2];
    const float* kb = (const float*)d_in[6 + 8 * l + 3];
    const float* vw = (const float*)d_in[6 + 8 * l + 4];
    const float* vb = (const float*)d_in[6 + 8 * l + 5];
    const float* ow = (const float*)d_in[6 + 8 * l + 6];
    const float* ob = (const float*)d_in[6 + 8 * l + 7];

    char* ws = (char*)d_ws;
    size_t qkBytes = (size_t)B * NHEADS * N * 64 * sizeof(__bf16);
    size_t vBytes  = (size_t)B * NHEADS * N * HEAD_DIM * sizeof(float);
    size_t aBytes  = (size_t)B * N * TOKEN_DIM * sizeof(__bf16);
    __bf16* Qbf = (__bf16*)ws;  ws += ((qkBytes + 255) / 256) * 256;
    __bf16* Kbf = (__bf16*)ws;  ws += ((qkBytes + 255) / 256) * 256;
    float*  Vbf = (float*)ws;   ws += ((vBytes + 255) / 256) * 256;
    __bf16* aOut = (__bf16*)ws; ws += ((aBytes + 255) / 256) * 256;

    // zero the K-dim pad (48..63) of Q/K rows
    hipMemsetAsync(Qbf, 0, qkBytes, stream);
    hipMemsetAsync(Kbf, 0, qkBytes, stream);

    dim3 gp(N / 16, TOKEN_DIM / 16, B);
    proj_kernel<<<gp, 32, 0, stream>>>(feat1, qw, qb, (void*)Qbf, N, C, W, 0);
    proj_kernel<<<gp, 32, 0, stream>>>(feat2, kw, kb, (void*)Kbf, N, C, W, 1);
    proj_kernel<<<gp, 32, 0, stream>>>(feat2, vw, vb, (void*)Vbf, N, C, W, 2);

    size_t smem = (size_t)16 * N * sizeof(float) +
                  (size_t)AWAVES * TOPK * 8 + AWAVES * 4 +
                  (size_t)AWAVES * 256 * 4 + 64;
    attn_kernel<<<dim3(N / 16, NHEADS, B), 32 * AWAVES, smem, stream>>>(
        Qbf, Kbf, Vbf, aOut, N, 1.0f / sqrtf((float)HEAD_DIM));

    outproj_kernel<<<gp, 32, 0, stream>>>(aOut, ow, ob, outPtr, N);
    outPtr += (size_t)B * TOKEN_DIM * N;
  }
}
